// Attention_61452392071637
// MI455X (gfx1250) — matrix operand order repaired and checked by reading, not compiled, not run on hardware
//
#include <hip/hip_runtime.h>
#include <hip/hip_bf16.h>

// ---------------------------------------------------------------------------
// Multi-head attention forward for MI455X (gfx1250), wave32 + WMMA bf16.
// B=4, S=2048, D=1024, H=16, HD=64.
// - All WMMA fragments load as contiguous b128 (A row-major, B transposed).
// - Attention stages K / V^T chunks into LDS via the Tensor Data Mover
//   (tensor_load_to_lds), double-buffered, wave-0 leader + s_wait_tensorcnt,
//   shared by all 8 waves of the block (same (b,h)).
// ---------------------------------------------------------------------------

#define B_    4
#define S_    2048
#define D_    1024
#define H_    16
#define HD_   64
#define MTOT  (B_ * S_)          // 8192 rows

typedef __attribute__((ext_vector_type(16))) __bf16 v16bf;
typedef __attribute__((ext_vector_type(4)))  __bf16 v4bf;
typedef __attribute__((ext_vector_type(8)))  float  v8f;
typedef __attribute__((ext_vector_type(4)))  unsigned v4u;
typedef __attribute__((ext_vector_type(8)))  int    v8i;
typedef __attribute__((ext_vector_type(4)))  int    v4i;

// ------------------------- WMMA helpers ------------------------------------

__device__ __forceinline__ v8f wmma_bf16(v16bf a, v16bf b, v8f c) {
  return __builtin_amdgcn_wmma_f32_16x16x32_bf16(
      false, a, false, b, (short)0, c, false, false);
}

// A-matrix 16x32 bf16 fragment from row-major source (leading dim lda).
__device__ __forceinline__ v16bf load_a16x32(const __bf16* __restrict__ a, int lda) {
  int lane = threadIdx.x & 31;
  int row  = lane & 15;
  int hi   = lane >> 4;
  const __bf16* r = a + row * lda;
  union { v4u q[2]; v16bf v; } u;
  u.q[0] = *(const v4u*)(r + (hi ? 8 : 0));
  u.q[1] = *(const v4u*)(r + (hi ? 24 : 16));
  return u.v;
}

// B-matrix 32x16 bf16 fragment from TRANSPOSED storage:
// logical B element (k, n) lives at bt[n*ldbt + k].
__device__ __forceinline__ v16bf load_bT16(const __bf16* __restrict__ bt, int ldbt) {
  int lane = threadIdx.x & 31;
  int col  = lane & 15;
  int hi   = lane >> 4;
  const __bf16* r = bt + col * ldbt;
  union { v4u q[2]; v16bf v; } u;
  u.q[0] = *(const v4u*)(r + (hi ? 8 : 0));
  u.q[1] = *(const v4u*)(r + (hi ? 24 : 16));
  return u.v;
}

// ------------------------- TDM: 2D tile -> LDS ------------------------------
// Builds a D# per CDNA5 ISA 8.3/8.4 (count=1, type=2/image, data_size=2B,
// 2D tile, groups 2/3 zero) and issues TENSOR_LOAD_TO_LDS.
// All dims/strides in bf16-element units; stride0 = elements between rows.
// Toolchain here is the 6-arg builtin: (v4u, v8i, v4i, v4i, v8i, i32 cpol).
__device__ __forceinline__ void tdm_load_2d(unsigned lds_off, const void* gaddr,
                                            unsigned tensor_d0, unsigned tensor_d1,
                                            unsigned tile_d0, unsigned tile_d1,
                                            unsigned long long stride0) {
  unsigned long long ga = (unsigned long long)(size_t)gaddr;
  v4u g0;
  g0[0] = 1u;                                            // count=1 (user D#)
  g0[1] = lds_off;                                       // lds_addr (bytes)
  g0[2] = (unsigned)(ga & 0xFFFFFFFFu);                  // global_addr[31:0]
  g0[3] = (unsigned)((ga >> 32) & 0x01FFFFFFu) | (2u << 30);  // [56:32] | type=2
  v8i g1;
  g1[0] = (int)(1u << 16);                               // data_size=1 (2 bytes)
  g1[1] = (int)((tensor_d0 & 0xFFFFu) << 16);            // tensor_dim0[15:0]
  g1[2] = (int)(((tensor_d0 >> 16) & 0xFFFFu) |
                ((tensor_d1 & 0xFFFFu) << 16));          // d0[31:16] | d1[15:0]
  g1[3] = (int)(((tensor_d1 >> 16) & 0xFFFFu) |
                ((tile_d0 & 0xFFFFu) << 16));            // d1[31:16] | tile0
  g1[4] = (int)(tile_d1 & 0xFFFFu);                      // tile1 | tile2=0
  g1[5] = (int)(unsigned)(stride0 & 0xFFFFFFFFull);      // stride0[31:0]
  g1[6] = (int)(unsigned)((stride0 >> 32) & 0xFFFFull);  // stride0[47:32]
  g1[7] = 0;                                             // dim1_stride=0 (2D)
  v4i z4 = {0, 0, 0, 0};
  v8i z8 = {0, 0, 0, 0, 0, 0, 0, 0};
  __builtin_amdgcn_tensor_load_to_lds(g0, g1, z4, z4, z8, 0);
}

// ------------------------- conversions --------------------------------------

__global__ __launch_bounds__(256)
void cvt_f32_to_bf16_v4(const float* __restrict__ in,
                        __bf16* __restrict__ out, int n4) {
  int i = blockIdx.x * blockDim.x + threadIdx.x;
  if (i < n4) {
    float4 f = ((const float4*)in)[i];
    v4bf o;
    o[0] = (__bf16)f.x; o[1] = (__bf16)f.y;
    o[2] = (__bf16)f.z; o[3] = (__bf16)f.w;
    ((v4bf*)out)[i] = o;
  }
}

// W[D,D] f32 row-major -> Wt[D,D] bf16, Wt[n*D+k] = W[k*D+n].
__global__ __launch_bounds__(256)
void transpose_cvt(const float* __restrict__ in, __bf16* __restrict__ out) {
  __shared__ float tile[32][33];
  int tx = threadIdx.x, ty = threadIdx.y;
  int x = blockIdx.x * 32 + tx;
  int y = blockIdx.y * 32 + ty;
#pragma unroll
  for (int j = 0; j < 32; j += 8)
    tile[ty + j][tx] = in[(y + j) * D_ + x];
  __syncthreads();
  int x2 = blockIdx.y * 32 + tx;
  int y2 = blockIdx.x * 32 + ty;
#pragma unroll
  for (int j = 0; j < 32; j += 8)
    out[(y2 + j) * D_ + x2] = (__bf16)tile[tx][ty + j];
}

// ------------------------- projection GEMM ----------------------------------
// Out = X[MTOT,D] @ W[D,D] + bias (W supplied transposed), bf16 in, f32 acc.
// One wave -> 32(M) x 64(N); 8 WMMA per 32-wide k-step; 4096 waves.
// mode 0: head-split bf16 [B,H,S,HD]            (Q and K)
// mode 1: head-split transposed bf16 [B,H,HD,S] (V)
// mode 2: flat f32 [MTOT,D]                     (final output)
__global__ __launch_bounds__(256)
void gemm_proj(const __bf16* __restrict__ X,
               const __bf16* __restrict__ Wt,
               const float*  __restrict__ bias,
               void* __restrict__ out, int mode) {
  int wave  = threadIdx.x >> 5;
  int gwave = blockIdx.x * 8 + wave;
  int mt = gwave >> 4;
  int nt = gwave & 15;
  int m0 = mt * 32;
  int n0 = nt * 64;

  const __bf16* a0p = X + (size_t)m0 * D_;
  const __bf16* a1p = a0p + (size_t)16 * D_;

  v8f acc[2][4];
#pragma unroll
  for (int i = 0; i < 2; ++i)
#pragma unroll
    for (int j = 0; j < 4; ++j) acc[i][j] = v8f{};

  for (int k = 0; k < D_; k += 32) {
    if (k + 32 < D_) {
      __builtin_prefetch(a0p + k + 32, 0, 0);
      __builtin_prefetch(Wt + (size_t)n0 * D_ + k + 32, 0, 0);
    }
    v16bf a0 = load_a16x32(a0p + k, D_);
    v16bf a1 = load_a16x32(a1p + k, D_);
#pragma unroll
    for (int j = 0; j < 4; ++j) {
      v16bf b = load_bT16(Wt + (size_t)(n0 + j * 16) * D_ + k, D_);
      acc[0][j] = wmma_bf16(a0, b, acc[0][j]);
      acc[1][j] = wmma_bf16(a1, b, acc[1][j]);
    }
  }

  int lane = threadIdx.x & 31;
  int col  = lane & 15;
  int rb   = (lane >> 4) * 8;
#pragma unroll
  for (int j = 0; j < 4; ++j) {
    int n  = n0 + j * 16 + col;
    float bv = bias[n];
    int h  = n >> 6;
    int hd = n & 63;
#pragma unroll
    for (int mi = 0; mi < 2; ++mi) {
#pragma unroll
      for (int r = 0; r < 8; ++r) {
        int m = m0 + mi * 16 + rb + r;
        int b = m >> 11;
        int s = m & 2047;
        float val = acc[mi][j][r] + bv;
        if (mode == 0) {
          ((__bf16*)out)[((size_t)(b * H_ + h) * S_ + s) * HD_ + hd] = (__bf16)val;
        } else if (mode == 1) {
          ((__bf16*)out)[((size_t)(b * H_ + h) * HD_ + hd) * S_ + s] = (__bf16)val;
        } else {
          ((float*)out)[(size_t)m * D_ + n] = val;
        }
      }
    }
  }
}

// ------------------------- flash attention core ------------------------------
// All 8 waves of a block share one (b,h); K / V^T chunks are staged into LDS
// once per block by the TDM (wave-0 leader, double-buffered), then every wave
// reads its B-fragments from LDS (ds_load_b128).
__global__ __launch_bounds__(256)
void attn_fwd(const __bf16* __restrict__ Q,    // [B,H,S,HD]
              const __bf16* __restrict__ K,    // [B,H,S,HD]
              const __bf16* __restrict__ Vt,   // [B,H,HD,S]
              __bf16* __restrict__ Aout) {     // [B,S,D] head-merged
  __shared__ __attribute__((aligned(16))) __bf16 ksmem[2][32 * HD_]; // [key][hd]
  __shared__ __attribute__((aligned(16))) __bf16 vsmem[2][HD_ * 32]; // [hd][key]
  __shared__ __attribute__((aligned(16))) __bf16 lds_p[8][16 * 32];  // per-wave P

  int wave  = threadIdx.x >> 5;
  int lane  = threadIdx.x & 31;
  int gwave = blockIdx.x * 8 + wave;           // 0..8191
  int qt = gwave & 127;
  int bh = gwave >> 7;                         // same for all waves in block
  int h  = bh & (H_ - 1);
  int b  = bh >> 4;
  int q0 = qt * 16;

  const __bf16* Qp = Q  + ((size_t)bh * S_ + q0) * HD_;
  const __bf16* Kp = K  + (size_t)bh * S_ * HD_;
  const __bf16* Vp = Vt + (size_t)bh * HD_ * S_;

  v16bf aq0 = load_a16x32(Qp, HD_);
  v16bf aq1 = load_a16x32(Qp + 32, HD_);

  int col  = lane & 15;
  int half = lane >> 4;
  int rb   = half * 8;

  float mrow[8], lrow[8];
#pragma unroll
  for (int r = 0; r < 8; ++r) { mrow[r] = -1e30f; lrow[r] = 0.f; }
  v8f o[4] = {v8f{}, v8f{}, v8f{}, v8f{}};
  const float scale = 0.125f;

  // scalar (SGPR) leader test so non-leader waves branch over the TDM issue
  bool leader = (__builtin_amdgcn_readfirstlane(threadIdx.x) >> 5) == 0;

  unsigned ksoff[2] = { (unsigned)(size_t)(void*)&ksmem[0][0],
                        (unsigned)(size_t)(void*)&ksmem[1][0] };
  unsigned vsoff[2] = { (unsigned)(size_t)(void*)&vsmem[0][0],
                        (unsigned)(size_t)(void*)&vsmem[1][0] };

  const int NCH = S_ / 32;                     // 64 key chunks
  if (leader) {
    // chunk 0 -> buffer 0
    tdm_load_2d(ksoff[0], Kp, HD_, S_, HD_, 32, HD_);
    tdm_load_2d(vsoff[0], Vp, S_, HD_, 32, HD_, S_);
  }

  for (int ic = 0; ic < NCH; ++ic) {
    int kc = ic * 32;
    int cb = ic & 1;
    if (leader) {
      if (ic + 1 < NCH) {
        int kn = kc + 32;
        tdm_load_2d(ksoff[cb ^ 1], Kp + (size_t)kn * HD_, HD_, S_, HD_, 32, HD_);
        tdm_load_2d(vsoff[cb ^ 1], Vp + kn, S_, HD_, 32, HD_, S_);
        __builtin_amdgcn_s_wait_tensorcnt(2);  // current pair complete
      } else {
        __builtin_amdgcn_s_wait_tensorcnt(0);
      }
    }
    __syncthreads();                           // buffer cb ready

    const __bf16* ks = &ksmem[cb][0];          // [32 keys][64 hd]
    const __bf16* vs = &vsmem[cb][0];          // [64 hd][32 keys]

    // scores: S[16x32] = Q(16x64) @ K^T ; B-frags from LDS (transposed store)
    v8f s0 = v8f{}, s1 = v8f{};
    {
      v16bf bk;
      bk = load_bT16(ks, HD_);                 s0 = wmma_bf16(aq0, bk, s0);
      bk = load_bT16(ks + 32, HD_);            s0 = wmma_bf16(aq1, bk, s0);
      bk = load_bT16(ks + 16 * HD_, HD_);      s1 = wmma_bf16(aq0, bk, s1);
      bk = load_bT16(ks + 16 * HD_ + 32, HD_); s1 = wmma_bf16(aq1, bk, s1);
    }
    // online softmax over this 32-key chunk
#pragma unroll
    for (int r = 0; r < 8; ++r) {
      float v0 = s0[r] * scale;
      float v1 = s1[r] * scale;
      float mx = fmaxf(v0, v1);
      mx = fmaxf(mx, __shfl_xor(mx, 1));
      mx = fmaxf(mx, __shfl_xor(mx, 2));
      mx = fmaxf(mx, __shfl_xor(mx, 4));
      mx = fmaxf(mx, __shfl_xor(mx, 8));
      float mnew = fmaxf(mrow[r], mx);
      float corr = __expf(mrow[r] - mnew);
      mrow[r] = mnew;
      float p0 = __expf(v0 - mnew);
      float p1 = __expf(v1 - mnew);
      float ps = p0 + p1;
      ps += __shfl_xor(ps, 1);
      ps += __shfl_xor(ps, 2);
      ps += __shfl_xor(ps, 4);
      ps += __shfl_xor(ps, 8);
      lrow[r] = lrow[r] * corr + ps;
      o[0][r] *= corr; o[1][r] *= corr; o[2][r] *= corr; o[3][r] *= corr;
      int row = rb + r;
      lds_p[wave][row * 32 + col]      = (__bf16)p0;
      lds_p[wave][row * 32 + 16 + col] = (__bf16)p1;
    }
    // O += P(16x32) @ V(32x64); V^T in LDS -> contiguous B-frag loads
    v16bf pa = load_a16x32(&lds_p[wave][0], 32);
#pragma unroll
    for (int j = 0; j < 4; ++j) {
      v16bf bv = load_bT16(vs + (j * 16) * 32, 32);
      o[j] = wmma_bf16(pa, bv, o[j]);
    }
    __syncthreads();                           // done reading buffer cb
  }

  __bf16* Ap = Aout + (size_t)(b * S_ + q0) * D_ + h * HD_;
#pragma unroll
  for (int r = 0; r < 8; ++r) {
    float inv = 1.0f / lrow[r];
    int row = rb + r;
#pragma unroll
    for (int j = 0; j < 4; ++j) {
      Ap[(size_t)row * D_ + j * 16 + col] = (__bf16)(o[j][r] * inv);
    }
  }
}

// ---------------------------------------------------------------------------

extern "C" void kernel_launch(void* const* d_in, const int* in_sizes, int n_in,
                              void* d_out, int out_size, void* d_ws, size_t ws_size,
                              hipStream_t stream) {
  const float* x  = (const float*)d_in[0];
  const float* Wq = (const float*)d_in[1];
  const float* bq = (const float*)d_in[2];
  const float* Wk = (const float*)d_in[3];
  const float* bk = (const float*)d_in[4];
  const float* Wv = (const float*)d_in[5];
  const float* bv = (const float*)d_in[6];
  const float* Wo = (const float*)d_in[7];
  const float* bo = (const float*)d_in[8];
  float* out = (float*)d_out;

  const size_t MB = 1024ull * 1024ull;
  char* ws = (char*)d_ws;
  __bf16* Xb   = (__bf16*)(ws + 0);          // 16 MB  [MTOT, D]
  __bf16* WqT  = (__bf16*)(ws + 16 * MB);    //  2 MB each, W^T bf16
  __bf16* WkT  = (__bf16*)(ws + 18 * MB);
  __bf16* WvT  = (__bf16*)(ws + 20 * MB);
  __bf16* WoT  = (__bf16*)(ws + 22 * MB);
  __bf16* Qb   = (__bf16*)(ws + 24 * MB);    // 16 MB  [B,H,S,HD]
  __bf16* Kb   = (__bf16*)(ws + 40 * MB);    // 16 MB  [B,H,S,HD]
  __bf16* Vtb  = (__bf16*)(ws + 56 * MB);    // 16 MB  [B,H,HD,S]
  __bf16* Ab   = (__bf16*)(ws + 72 * MB);    // 16 MB  [B,S,D]

  {
    int n4 = (MTOT * D_) / 4;
    cvt_f32_to_bf16_v4<<<n4 / 256, 256, 0, stream>>>(x, Xb, n4);
    dim3 tg(32, 32), tb(32, 8);
    transpose_cvt<<<tg, tb, 0, stream>>>(Wq, WqT);
    transpose_cvt<<<tg, tb, 0, stream>>>(Wk, WkT);
    transpose_cvt<<<tg, tb, 0, stream>>>(Wv, WvT);
    transpose_cvt<<<tg, tb, 0, stream>>>(Wo, WoT);
  }

  gemm_proj<<<512, 256, 0, stream>>>(Xb, WqT, bq, (void*)Qb, 0);
  gemm_proj<<<512, 256, 0, stream>>>(Xb, WkT, bk, (void*)Kb, 0);
  gemm_proj<<<512, 256, 0, stream>>>(Xb, WvT, bv, (void*)Vtb, 1);

  attn_fwd<<<1024, 256, 0, stream>>>(Qb, Kb, Vtb, Ab);

  gemm_proj<<<512, 256, 0, stream>>>(Ab, WoT, bo, (void*)out, 2);
}
